// ChunkedLinearAttention_59158879535755
// MI455X (gfx1250) — matrix operand order repaired and checked by reading, not compiled, not run on hardware
//
#include <hip/hip_runtime.h>

typedef _Float16 v8h  __attribute__((ext_vector_type(8)));
typedef _Float16 v16h __attribute__((ext_vector_type(16)));
typedef float    v8f  __attribute__((ext_vector_type(8)));

#define DIM_   1024
#define H_     16
#define D_     64
#define CS_    64
#define NC_    64
#define BATCH_ 4
#define SEQ_   4096
#define ROWS_  (BATCH_ * SEQ_)   // 16384
#define QKV_   (3 * DIM_)        // 3072
#define SCALE_ 0.125f            // D^-0.5

// ---------------------------------------------------------------------------
// CDNA5 async global->LDS copy (ASYNCcnt-tracked DMA, no VGPR round trip).
// VDST operand is the LDS byte offset: generic shared pointers map their low
// 32 bits straight onto the LDS offset (ISA 10.2 aperture rules).
// ---------------------------------------------------------------------------

__device__ __forceinline__ void async_b128(const _Float16* g, _Float16* l) {
    unsigned lds = (unsigned)(size_t)l;
    asm volatile("global_load_async_to_lds_b128 %0, %1, off"
                 :: "v"(lds), "v"(g) : "memory");
}

__device__ __forceinline__ void wait_async0() {
    asm volatile("s_wait_asynccnt 0" ::: "memory");
}

// ---------------------------------------------------------------------------
// WMMA fragment helpers (wave32, v_wmma_f32_16x16x32_f16)
// A 16x32 f16 layout: lane L<16 -> row M=L, K = {k0..k0+7, k0+16..k0+23}
//                     lane L>=16 -> row M=L-16, K = {k0+8..k0+15, k0+24..k0+31}
// B 32x16 f16 layout: lane L<16 -> col N=L, K = {k0..k0+7, k0+16..k0+23}
//                     lane L>=16 -> col N=L-16, K = {k0+8..k0+15, k0+24..k0+31}
// C/D 16x16 f32: VGPR v, lane L: (M = v + (L>=16 ? 8 : 0), N = L%16)
// ---------------------------------------------------------------------------

__device__ __forceinline__ v16h make16(v8h lo, v8h hi) {
    return __builtin_shufflevector(lo, hi, 0, 1, 2, 3, 4, 5, 6, 7,
                                           8, 9, 10, 11, 12, 13, 14, 15);
}

__device__ __forceinline__ v16h frag_a(const _Float16* base, int ld, int row0,
                                       int k0, int lane) {
    int r  = row0 + (lane & 15);
    int kb = k0 + ((lane & 16) >> 1);           // +8 for hi half
    const _Float16* p = base + r * ld + kb;
    v8h lo = *(const v8h*)p;
    v8h hi = *(const v8h*)(p + 16);
    return make16(lo, hi);
}

__device__ __forceinline__ v16h frag_b(const _Float16* base, int ld, int col0,
                                       int k0, int lane) {
    int n  = col0 + (lane & 15);
    int kb = k0 + ((lane & 16) >> 1);           // +8 for hi half
    const _Float16* p = base + n * ld + kb;
    v8h lo = *(const v8h*)p;
    v8h hi = *(const v8h*)(p + 16);
    return make16(lo, hi);
}

__device__ __forceinline__ v8f wmma_f16(v16h a, v16h b, v8f c) {
    return __builtin_amdgcn_wmma_f32_16x16x32_f16(
        /*neg_a=*/false, a, /*neg_b=*/false, b,
        /*c_mod=*/(short)0, c, /*reuse_a=*/false, /*reuse_b=*/false);
}

// ---------------------------------------------------------------------------
// Precision-conversion kernels
// ---------------------------------------------------------------------------

__global__ void convert_f16_kernel(const float* __restrict__ src,
                                   _Float16* __restrict__ dst, size_t total) {
    size_t idx = (size_t)blockIdx.x * blockDim.x + threadIdx.x;
    if (idx < total) dst[idx] = (_Float16)src[idx];
}

// src [K][N] f32 (row-major) -> dst [N][K] f16
__global__ void transpose_f16_kernel(const float* __restrict__ src,
                                     _Float16* __restrict__ dst, int K, int N) {
    size_t idx = (size_t)blockIdx.x * blockDim.x + threadIdx.x;
    size_t total = (size_t)K * N;
    if (idx >= total) return;
    int n = (int)(idx % N);
    int k = (int)(idx / N);
    dst[(size_t)n * K + k] = (_Float16)src[idx];
}

// ---------------------------------------------------------------------------
// Shared GEMM body: 128x128 workgroup tile, 8 waves (2 M-frags x 4 N-frags
// per wave), K-step 32, double-buffered LDS fed by async global->LDS DMA.
// ---------------------------------------------------------------------------

#define LDK 40   // 32 + 8 halves pad (keeps 16B alignment, spreads banks)

__device__ __forceinline__ void gemm_tile_async(const _Float16* __restrict__ A,
                                                const _Float16* __restrict__ Bt,
                                                _Float16* As, _Float16* Bs,
                                                int m_blk, int n_blk, int k0,
                                                int tid) {
#pragma unroll
    for (int it = 0; it < 2; ++it) {
        int idx = tid + it * 256;
        int row = idx >> 2;
        int seg = (idx & 3) << 3;
        async_b128(&A[(size_t)(m_blk + row) * DIM_ + k0 + seg],
                   &As[row * LDK + seg]);
        async_b128(&Bt[(size_t)(n_blk + row) * DIM_ + k0 + seg],
                   &Bs[row * LDK + seg]);
    }
}

__device__ __forceinline__ void gemm_core(const _Float16* __restrict__ A,
                                          const _Float16* __restrict__ Bt,
                                          _Float16 (*As)[128 * LDK],
                                          _Float16 (*Bs)[128 * LDK],
                                          v8f acc[2][4], int m_blk, int n_blk,
                                          int tid, int lane, int wm, int wn) {
    gemm_tile_async(A, Bt, As[0], Bs[0], m_blk, n_blk, 0, tid);
    wait_async0();
    __syncthreads();

    for (int k0 = 0; k0 < DIM_; k0 += 32) {
        const int cur = (k0 >> 5) & 1;
        const int nxt = cur ^ 1;
        if (k0 + 32 < DIM_)
            gemm_tile_async(A, Bt, As[nxt], Bs[nxt], m_blk, n_blk, k0 + 32, tid);

        v16h a0 = frag_a(As[cur], LDK, wm * 32, 0, lane);
        v16h a1 = frag_a(As[cur], LDK, wm * 32 + 16, 0, lane);
#pragma unroll
        for (int nf = 0; nf < 4; ++nf) {
            v16h bf = frag_b(Bs[cur], LDK, wn * 64 + nf * 16, 0, lane);
            acc[0][nf] = wmma_f16(a0, bf, acc[0][nf]);
            acc[1][nf] = wmma_f16(a1, bf, acc[1][nf]);
        }
        wait_async0();
        __syncthreads();
    }
}

// ---------------------------------------------------------------------------
// GEMM1: qkv = x @ W_qkv. Scatters q (pre-scaled) as [b][h][c][r][d],
// k as [b][h][c][r][d], v transposed as [b][h][c][d][r].
// ---------------------------------------------------------------------------

__global__ __launch_bounds__(256) void gemm_qkv_kernel(
    const _Float16* __restrict__ A, const _Float16* __restrict__ Bt,
    _Float16* __restrict__ qh, _Float16* __restrict__ kh,
    _Float16* __restrict__ vh) {
    __shared__ __align__(16) _Float16 As[2][128 * LDK];
    __shared__ __align__(16) _Float16 Bs[2][128 * LDK];

    const int tid  = threadIdx.x;
    const int lane = tid & 31;
    const int wave = tid >> 5;
    const int wm   = wave >> 1;
    const int wn   = wave & 1;
    const int m_blk = blockIdx.y * 128;
    const int n_blk = blockIdx.x * 128;

    v8f acc[2][4] = {};
    gemm_core(A, Bt, As, Bs, acc, m_blk, n_blk, tid, lane, wm, wn);

    const int hi8 = (lane >> 4) << 3;
#pragma unroll
    for (int mf = 0; mf < 2; ++mf)
#pragma unroll
        for (int nf = 0; nf < 4; ++nf)
#pragma unroll
            for (int v = 0; v < 8; ++v) {
                int i = m_blk + wm * 32 + mf * 16 + hi8 + v;
                int j = n_blk + wn * 64 + nf * 16 + (lane & 15);
                float val = acc[mf][nf][v];
                int b = i >> 12, n = i & 4095;
                int c = n >> 6,  r = n & 63;
                int which = j >> 10, rem = j & 1023;
                int h = rem >> 6, d = rem & 63;
                size_t chunk = ((size_t)((b * H_ + h) * NC_ + c)) * (CS_ * D_);
                if (which == 0)      qh[chunk + r * D_ + d] = (_Float16)(val * SCALE_);
                else if (which == 1) kh[chunk + r * D_ + d] = (_Float16)val;
                else                 vh[chunk + d * CS_ + r] = (_Float16)val;
            }
}

// ---------------------------------------------------------------------------
// Chunk statistics: exclusive cumsum of per-chunk K/V means (f32).
// ---------------------------------------------------------------------------

__global__ void chunk_stats_kernel(const _Float16* __restrict__ kh,
                                   const _Float16* __restrict__ vh,
                                   float* __restrict__ cumk,
                                   float* __restrict__ cumv) {
    int bh = blockIdx.x;    // 0..63
    int d  = threadIdx.x;   // 0..63
    float sk = 0.f, sv = 0.f;
    for (int c = 0; c < NC_; ++c) {
        size_t base = ((size_t)bh * NC_ + c) * (CS_ * D_);
        size_t sidx = ((size_t)bh * NC_ + c) * D_ + d;
        cumk[sidx] = sk;    // exclusive prefix
        cumv[sidx] = sv;
        float tk = 0.f, tv = 0.f;
        for (int r = 0; r < CS_; ++r) {
            tk += (float)kh[base + (size_t)r * D_ + d];
            tv += (float)vh[base + (size_t)d * CS_ + r];
        }
        sk += tk * (1.f / CS_);
        sv += tv * (1.f / CS_);
    }
}

// ---------------------------------------------------------------------------
// Per-chunk attention: one workgroup per (b,h,c), 128 threads (4 waves).
// Wave w owns output rows [w*16, w*16+16).
// ---------------------------------------------------------------------------

#define LDC 72   // 64 + 8 halves pad

__global__ __launch_bounds__(128) void attn_kernel(
    const _Float16* __restrict__ qh, const _Float16* __restrict__ kh,
    const _Float16* __restrict__ vh, const float* __restrict__ cumk,
    const float* __restrict__ cumv, _Float16* __restrict__ attnh) {
    __shared__ __align__(16) _Float16 qs[CS_ * LDC];
    __shared__ __align__(16) _Float16 ks[CS_ * LDC];
    __shared__ __align__(16) _Float16 vs[CS_ * LDC];
    __shared__ __align__(16) _Float16 ps[CS_ * LDC];
    __shared__ float cks[D_], cvs[D_], crs[CS_];

    const int blk = blockIdx.x;         // ((b*H + h)*NC + c)
    const int c   = blk & (NC_ - 1);
    const int bh  = blk >> 6;
    const int b   = bh >> 4;
    const int h   = bh & 15;
    const int tid  = threadIdx.x;
    const int lane = tid & 31;
    const int wave = tid >> 5;
    const size_t gbase = (size_t)blk * (CS_ * D_);

#pragma unroll
    for (int it = 0; it < 4; ++it) {
        int idx = tid + it * 128;
        int row = idx >> 3;
        int seg = (idx & 7) << 3;
        async_b128(&qh[gbase + row * D_ + seg], &qs[row * LDC + seg]);
        async_b128(&kh[gbase + row * D_ + seg], &ks[row * LDC + seg]);
        async_b128(&vh[gbase + row * CS_ + seg], &vs[row * LDC + seg]);
    }
    if (tid < D_) {
        cks[tid] = cumk[(size_t)blk * D_ + tid];
        cvs[tid] = cumv[(size_t)blk * D_ + tid];
    }
    wait_async0();
    __syncthreads();

    // cross gate per query row (q already pre-scaled by SCALE)
    if (tid < CS_) {
        float acc = 0.f;
        for (int d = 0; d < D_; ++d) acc += (float)qs[tid * LDC + d] * cks[d];
        crs[tid] = 1.f / (1.f + __expf(-acc));
    }

    // S = q @ k^T  (already scaled)
    const int m0 = wave * 16;
    v8f s[4] = {};
#pragma unroll
    for (int k0 = 0; k0 < D_; k0 += 32) {
        v16h a = frag_a(qs, LDC, m0, k0, lane);
#pragma unroll
        for (int nf = 0; nf < 4; ++nf)
            s[nf] = wmma_f16(a, frag_b(ks, LDC, nf * 16, k0, lane), s[nf]);
    }

    // causal mask + row softmax (row = VGPR index; 16 lanes hold the columns)
    const int hi8  = (lane >> 4) << 3;
    const int ncol = lane & 15;
    float rmax[8], rsum[8];
#pragma unroll
    for (int v = 0; v < 8; ++v) rmax[v] = -3.0e38f;
#pragma unroll
    for (int nf = 0; nf < 4; ++nf)
#pragma unroll
        for (int v = 0; v < 8; ++v) {
            int m = m0 + hi8 + v;
            int n = nf * 16 + ncol;
            float val = s[nf][v];
            if (n > m) val = -65000.f;
            s[nf][v] = val;
            rmax[v] = fmaxf(rmax[v], val);
        }
#pragma unroll
    for (int v = 0; v < 8; ++v) {
        for (int off = 1; off < 16; off <<= 1)
            rmax[v] = fmaxf(rmax[v], __shfl_xor(rmax[v], off, 16));
        rsum[v] = 0.f;
    }
#pragma unroll
    for (int nf = 0; nf < 4; ++nf)
#pragma unroll
        for (int v = 0; v < 8; ++v) {
            float p = __expf(s[nf][v] - rmax[v]);
            s[nf][v] = p;
            rsum[v] += p;
        }
#pragma unroll
    for (int v = 0; v < 8; ++v) {
        for (int off = 1; off < 16; off <<= 1)
            rsum[v] += __shfl_xor(rsum[v], off, 16);
        rsum[v] = 1.f / rsum[v];
    }
#pragma unroll
    for (int nf = 0; nf < 4; ++nf)
#pragma unroll
        for (int v = 0; v < 8; ++v) {
            int m = m0 + hi8 + v;
            int n = nf * 16 + ncol;
            ps[m * LDC + n] = (_Float16)(s[nf][v] * rsum[v]);
        }
    __syncthreads();

    // out_local = P @ v   (vs holds v transposed: [d][r] => B layout [col][K])
    v8f o[4] = {};
#pragma unroll
    for (int k0 = 0; k0 < CS_; k0 += 32) {
        v16h a = frag_a(ps, LDC, m0, k0, lane);
#pragma unroll
        for (int nf = 0; nf < 4; ++nf)
            o[nf] = wmma_f16(a, frag_b(vs, LDC, nf * 16, k0, lane), o[nf]);
    }

    // + cross * cum_v * 0.5; store f16 into [B,N,C]-layout activation buffer
#pragma unroll
    for (int nf = 0; nf < 4; ++nf)
#pragma unroll
        for (int v = 0; v < 8; ++v) {
            int m = m0 + hi8 + v;
            int d = nf * 16 + ncol;
            float val = o[nf][v] + crs[m] * cvs[d] * 0.5f;
            int i = b * SEQ_ + c * CS_ + m;
            int j = h * D_ + d;
            attnh[(size_t)i * DIM_ + j] = (_Float16)val;
        }
}

// ---------------------------------------------------------------------------
// GEMM2: out = attn @ W_out  (f16 x f16 -> f32 directly to d_out)
// ---------------------------------------------------------------------------

__global__ __launch_bounds__(256) void gemm_out_kernel(
    const _Float16* __restrict__ A, const _Float16* __restrict__ Bt,
    float* __restrict__ out) {
    __shared__ __align__(16) _Float16 As[2][128 * LDK];
    __shared__ __align__(16) _Float16 Bs[2][128 * LDK];

    const int tid  = threadIdx.x;
    const int lane = tid & 31;
    const int wave = tid >> 5;
    const int wm   = wave >> 1;
    const int wn   = wave & 1;
    const int m_blk = blockIdx.y * 128;
    const int n_blk = blockIdx.x * 128;

    v8f acc[2][4] = {};
    gemm_core(A, Bt, As, Bs, acc, m_blk, n_blk, tid, lane, wm, wn);

    const int hi8 = (lane >> 4) << 3;
#pragma unroll
    for (int mf = 0; mf < 2; ++mf)
#pragma unroll
        for (int nf = 0; nf < 4; ++nf)
#pragma unroll
            for (int v = 0; v < 8; ++v) {
                int i = m_blk + wm * 32 + mf * 16 + hi8 + v;
                int j = n_blk + wn * 64 + nf * 16 + (lane & 15);
                out[(size_t)i * DIM_ + j] = acc[mf][nf][v];
            }
}

// ---------------------------------------------------------------------------
// Launcher
// ---------------------------------------------------------------------------

extern "C" void kernel_launch(void* const* d_in, const int* in_sizes, int n_in,
                              void* d_out, int out_size, void* d_ws,
                              size_t ws_size, hipStream_t stream) {
    (void)in_sizes; (void)n_in; (void)out_size; (void)ws_size;
    const float* x    = (const float*)d_in[0];
    const float* Wqkv = (const float*)d_in[1];
    const float* Wout = (const float*)d_in[2];
    float* out = (float*)d_out;

    char* ws = (char*)d_ws;
    size_t off = 0;
    _Float16* xh    = (_Float16*)(ws + off); off += (size_t)ROWS_ * DIM_ * 2;
    _Float16* Wqkvt = (_Float16*)(ws + off); off += (size_t)QKV_ * DIM_ * 2;
    _Float16* Woutt = (_Float16*)(ws + off); off += (size_t)DIM_ * DIM_ * 2;
    _Float16* qh    = (_Float16*)(ws + off); off += (size_t)ROWS_ * DIM_ * 2;
    _Float16* kh    = (_Float16*)(ws + off); off += (size_t)ROWS_ * DIM_ * 2;
    _Float16* vh    = (_Float16*)(ws + off); off += (size_t)ROWS_ * DIM_ * 2;
    _Float16* attnh = (_Float16*)(ws + off); off += (size_t)ROWS_ * DIM_ * 2;
    float* cumk = (float*)(ws + off); off += (size_t)BATCH_ * H_ * NC_ * D_ * 4;
    float* cumv = (float*)(ws + off); off += (size_t)BATCH_ * H_ * NC_ * D_ * 4;

    size_t nx = (size_t)ROWS_ * DIM_;
    convert_f16_kernel<<<(int)((nx + 255) / 256), 256, 0, stream>>>(x, xh, nx);
    transpose_f16_kernel<<<(DIM_ * QKV_) / 256, 256, 0, stream>>>(Wqkv, Wqkvt, DIM_, QKV_);
    transpose_f16_kernel<<<(DIM_ * DIM_) / 256, 256, 0, stream>>>(Wout, Woutt, DIM_, DIM_);

    gemm_qkv_kernel<<<dim3(QKV_ / 128, ROWS_ / 128), 256, 0, stream>>>(
        xh, Wqkvt, qh, kh, vh);

    chunk_stats_kernel<<<BATCH_ * H_, 64, 0, stream>>>(kh, vh, cumk, cumv);

    attn_kernel<<<BATCH_ * H_ * NC_, 128, 0, stream>>>(qh, kh, vh, cumk, cumv,
                                                       attnh);

    gemm_out_kernel<<<dim3(DIM_ / 128, ROWS_ / 128), 256, 0, stream>>>(
        attnh, Woutt, out);
}
